// Attention_53704271069274
// MI455X (gfx1250) — matrix operand order repaired and checked by reading, not compiled, not run on hardware
//
#include <hip/hip_runtime.h>
#include <hip/hip_bf16.h>
#include <math.h>

#define DIM   768
#define HEADS 12
#define HD    64
#define BATCH 4
#define SEQ   2048
#define NTOK  (BATCH*SEQ)   // 8192
#define R_LORA 8

typedef __bf16 bf16;
typedef bf16  v16bf __attribute__((ext_vector_type(16)));
typedef float v8f   __attribute__((ext_vector_type(8)));

union Frag16 { v16bf v; uint4 u[2]; };
union BF8    { bf16 b[8]; uint4 u; };

__device__ __forceinline__ bf16 f2bf(float f) {
  union { float f; unsigned u; } x; x.f = f;
  unsigned r = (x.u + 0x7FFFu + ((x.u >> 16) & 1u)) >> 16;   // round-to-nearest-even
  union { unsigned short s; bf16 b; } y; y.s = (unsigned short)r;
  return y.b;
}

// ---------------------------------------------------------------------------
// Fold LoRA into the base weight: Wd[m][k] = bf16( W[m][k] + sum_r Bl[m][r]*A[r][k] )
// grid = DIM blocks (one output row m), 256 threads, 3 elements per thread.
// ---------------------------------------------------------------------------
__global__ __launch_bounds__(256)
void prep_weight(const float* __restrict__ W, const float* __restrict__ A,
                 const float* __restrict__ Bl, bf16* __restrict__ Wd) {
  const int m = blockIdx.x;
  float bl[R_LORA];
#pragma unroll
  for (int r = 0; r < R_LORA; ++r) bl[r] = Bl[m * R_LORA + r];
#pragma unroll
  for (int e = 0; e < 3; ++e) {
    const int kk = threadIdx.x + e * 256;
    float acc = W[(size_t)m * DIM + kk];
#pragma unroll
    for (int r = 0; r < R_LORA; ++r) acc += bl[r] * A[(size_t)r * DIM + kk];
    Wd[(size_t)m * DIM + kk] = f2bf(acc);
  }
}

__global__ __launch_bounds__(256)
void copyf(const float* __restrict__ s, float* __restrict__ d, int n) {
  const int i = blockIdx.x * blockDim.x + threadIdx.x;
  if (i < n) d[i] = s[i];
}

// fp32 -> bf16, 8 elements per thread, vectorized 16B stores
__global__ __launch_bounds__(256)
void cvt_x(const float* __restrict__ X, bf16* __restrict__ Xb) {
  const int i = blockIdx.x * blockDim.x + threadIdx.x;
  const float4* p = reinterpret_cast<const float4*>(X) + (size_t)i * 2;
  const float4 a = p[0], b = p[1];
  BF8 o;
  o.b[0] = f2bf(a.x); o.b[1] = f2bf(a.y); o.b[2] = f2bf(a.z); o.b[3] = f2bf(a.w);
  o.b[4] = f2bf(b.x); o.b[5] = f2bf(b.y); o.b[6] = f2bf(b.z); o.b[7] = f2bf(b.w);
  reinterpret_cast<uint4*>(Xb)[i] = o.u;
}

// ---------------------------------------------------------------------------
// C[n,m] = sum_k X[n,k] * Wt[m,k] + bias[m]
// Workgroup tile: 64 tokens x 128 channels; 8 waves (2x4); wave tile 32x32.
// mode 0: scatter bf16 into q/k/v [b,h,t,64]; mode 1: f32 into outf[n*768+m].
// ---------------------------------------------------------------------------
__global__ __launch_bounds__(256)
void gemm_bf16(const bf16* __restrict__ X, const bf16* __restrict__ Wt,
               const float* __restrict__ bias, int chanTiles, int mode,
               bf16* __restrict__ qd, bf16* __restrict__ kd, bf16* __restrict__ vd,
               float* __restrict__ outf) {
  const int lane = threadIdx.x & 31;
  const int w    = threadIdx.x >> 5;
  const int wm = w >> 2, wn = w & 3;
  const int gx = blockIdx.x % chanTiles;
  const int gy = blockIdx.x / chanTiles;
  const int tokBase  = gy * 64  + wm * 32;
  const int chanBase = gx * 128 + wn * 32;
  const int lrow = lane & 15;
  const int hi   = (lane >> 4) & 1;
  const int c0   = hi ? 8 : 0;

  v8f acc[2][2] = {};

  for (int kk = 0; kk < DIM; kk += 32) {
    Frag16 a[2], b[2];
#pragma unroll
    for (int i = 0; i < 2; ++i) {
      const bf16* p = X + (size_t)(tokBase + i * 16 + lrow) * DIM + kk;
      a[i].u[0] = *reinterpret_cast<const uint4*>(p + c0);        // K (0..7 | 8..15)
      a[i].u[1] = *reinterpret_cast<const uint4*>(p + 16 + c0);   // K (16..23 | 24..31)
    }
#pragma unroll
    for (int j = 0; j < 2; ++j) {
      const bf16* p = Wt + (size_t)(chanBase + j * 16 + lrow) * DIM + kk;
      b[j].u[0] = *reinterpret_cast<const uint4*>(p + c0);        // K (0..7 | 8..15)
      b[j].u[1] = *reinterpret_cast<const uint4*>(p + 16 + c0);
    }
#pragma unroll
    for (int i = 0; i < 2; ++i)
#pragma unroll
      for (int j = 0; j < 2; ++j)
        acc[i][j] = __builtin_amdgcn_wmma_f32_16x16x32_bf16(
            false, a[i].v, false, b[j].v, (short)0, acc[i][j], false, false);
  }

#pragma unroll
  for (int j = 0; j < 2; ++j) {
    const int ch = chanBase + j * 16 + lrow;
    const float bv = bias[ch];
#pragma unroll
    for (int i = 0; i < 2; ++i) {
#pragma unroll
      for (int r = 0; r < 8; ++r) {
        const float val = acc[i][j][r] + bv;
        const int tok = tokBase + i * 16 + r + hi * 8;   // C layout: lanes 16-31 hold M=r+8
        if (mode == 0) {
          const int proj = ch / DIM, within = ch % DIM;
          const int head = within >> 6, off = within & 63;
          const int bb = tok >> 11, tq = tok & (SEQ - 1);
          bf16* d = (proj == 0) ? qd : ((proj == 1) ? kd : vd);
          d[((size_t)((bb * HEADS + head) * SEQ + tq)) * HD + off] = f2bf(val);
        } else {
          outf[(size_t)tok * DIM + ch] = val;
        }
      }
    }
  }
}

// ---------------------------------------------------------------------------
// Flash attention: block = 8 waves * 16 query rows = 128 rows of one (b,h).
// Keys processed in chunks of 32; online softmax per row (stats replicated
// across the 16-lane half holding that row of the C-layout tile).
// ---------------------------------------------------------------------------
__global__ __launch_bounds__(256)
void attn_kernel(const bf16* __restrict__ Q, const bf16* __restrict__ K,
                 const bf16* __restrict__ V, bf16* __restrict__ ctx) {
  __shared__ bf16 vt[HD * 32];        // V chunk transposed: [c][key]
  __shared__ bf16 pbuf[8][16 * 32];   // per-wave P tile (row-major)

  const int lane = threadIdx.x & 31;
  const int w    = threadIdx.x >> 5;
  const int bh     = blockIdx.x >> 4;        // b*HEADS + h
  const int rowBlk = blockIdx.x & 15;
  const int b = bh / HEADS, h = bh % HEADS;
  const int qBase = rowBlk * 128 + w * 16;
  const int lrow = lane & 15;
  const int hi   = (lane >> 4) & 1;
  const int c0   = hi ? 8 : 0;
  const float scale = 0.125f;                // 64^-0.5

  const bf16* qh = Q + (size_t)bh * SEQ * HD;
  const bf16* kh = K + (size_t)bh * SEQ * HD;
  const bf16* vh = V + (size_t)bh * SEQ * HD;

  // Q as two 16x32 A-fragments (hd=64)
  Frag16 qa[2];
#pragma unroll
  for (int kc = 0; kc < 2; ++kc) {
    const bf16* p = qh + (size_t)(qBase + lrow) * HD + kc * 32;
    qa[kc].u[0] = *reinterpret_cast<const uint4*>(p + c0);
    qa[kc].u[1] = *reinterpret_cast<const uint4*>(p + 16 + c0);
  }

  v8f o[4] = {};
  float mrow[8], lsum[8];
#pragma unroll
  for (int r = 0; r < 8; ++r) { mrow[r] = -1e30f; lsum[r] = 0.0f; }

  for (int s0 = 0; s0 < SEQ; s0 += 32) {
    // cooperative V-chunk transpose into LDS (32 keys x 64 channels)
    __syncthreads();
    {
      const int key = threadIdx.x >> 3;        // 0..31
      const int cb  = (threadIdx.x & 7) * 8;   // 0..56
      BF8 vv; vv.u = *reinterpret_cast<const uint4*>(vh + (size_t)(s0 + key) * HD + cb);
#pragma unroll
      for (int e = 0; e < 8; ++e) vt[(cb + e) * 32 + key] = vv.b[e];
    }
    __syncthreads();

    // S = Q * K^T : two 16x16 key tiles
    v8f sacc[2] = {};
#pragma unroll
    for (int kt = 0; kt < 2; ++kt) {
#pragma unroll
      for (int kc = 0; kc < 2; ++kc) {
        Frag16 kbf;
        const bf16* p = kh + (size_t)(s0 + kt * 16 + lrow) * HD + kc * 32;
        kbf.u[0] = *reinterpret_cast<const uint4*>(p + c0);
        kbf.u[1] = *reinterpret_cast<const uint4*>(p + 16 + c0);
        sacc[kt] = __builtin_amdgcn_wmma_f32_16x16x32_bf16(
            false, qa[kc].v, false, kbf.v, (short)0, sacc[kt], false, false);
      }
    }

    // online softmax, row r lives across the 16-lane half (lanes = key index)
#pragma unroll
    for (int r = 0; r < 8; ++r) {
      const float sA = sacc[0][r] * scale;
      const float sB = sacc[1][r] * scale;
      float mx = fmaxf(sA, sB);
#pragma unroll
      for (int m = 1; m <= 8; m <<= 1) mx = fmaxf(mx, __shfl_xor(mx, m, 32));
      const float mnew  = fmaxf(mrow[r], mx);
      const float alpha = __expf(mrow[r] - mnew);
      const float p0 = __expf(sA - mnew);
      const float p1 = __expf(sB - mnew);
      float rs = p0 + p1;
#pragma unroll
      for (int m = 1; m <= 8; m <<= 1) rs += __shfl_xor(rs, m, 32);
      lsum[r] = lsum[r] * alpha + rs;
      mrow[r] = mnew;
#pragma unroll
      for (int j = 0; j < 4; ++j) o[j][r] *= alpha;
      const int prow = r + hi * 8;
      pbuf[w][prow * 32 + lrow]      = f2bf(p0);
      pbuf[w][prow * 32 + 16 + lrow] = f2bf(p1);
    }
    __builtin_amdgcn_wave_barrier();

    // reload P in A-fragment layout (16x32 over 32 keys)
    Frag16 pa;
    {
      const bf16* p = &pbuf[w][lrow * 32];
      pa.u[0] = *reinterpret_cast<const uint4*>(p + c0);
      pa.u[1] = *reinterpret_cast<const uint4*>(p + 16 + c0);
    }
    // O += P * V, 4 column tiles of 16
#pragma unroll
    for (int j = 0; j < 4; ++j) {
      Frag16 vbf;
      const bf16* p = &vt[(j * 16 + lrow) * 32];
      vbf.u[0] = *reinterpret_cast<const uint4*>(p + c0);
      vbf.u[1] = *reinterpret_cast<const uint4*>(p + 16 + c0);
      o[j] = __builtin_amdgcn_wmma_f32_16x16x32_bf16(
          false, pa.v, false, vbf.v, (short)0, o[j], false, false);
    }
  }

  // normalize and store context in [token][dim] bf16 layout
#pragma unroll
  for (int r = 0; r < 8; ++r) {
    const float inv = 1.0f / lsum[r];
    const int tok = qBase + r + hi * 8;
    const size_t base = ((size_t)(b * SEQ + tok)) * DIM + h * HD;
#pragma unroll
    for (int j = 0; j < 4; ++j)
      ctx[base + j * 16 + lrow] = f2bf(o[j][r] * inv);
  }
}

// ---------------------------------------------------------------------------
extern "C" void kernel_launch(void* const* d_in, const int* in_sizes, int n_in,
                              void* d_out, int out_size, void* d_ws, size_t ws_size,
                              hipStream_t stream) {
  const float* x   = (const float*)d_in[0];
  const float* wq  = (const float*)d_in[1];
  const float* bq  = (const float*)d_in[2];
  const float* laq = (const float*)d_in[3];
  const float* lbq = (const float*)d_in[4];
  const float* wk  = (const float*)d_in[5];
  const float* bk  = (const float*)d_in[6];
  const float* lak = (const float*)d_in[7];
  const float* lbk = (const float*)d_in[8];
  const float* wv  = (const float*)d_in[9];
  const float* bv  = (const float*)d_in[10];
  const float* lav = (const float*)d_in[11];
  const float* lbv = (const float*)d_in[12];
  const float* wo  = (const float*)d_in[13];
  const float* bo  = (const float*)d_in[14];
  const float* lao = (const float*)d_in[15];
  const float* lbo = (const float*)d_in[16];
  float* out = (float*)d_out;
  (void)in_sizes; (void)n_in; (void)out_size; (void)ws_size;

  char* ws = (char*)d_ws;
  size_t off = 0;
  auto alloc = [&](size_t bytes) -> void* {
    void* p = ws + off;
    off = (off + bytes + 255) & ~(size_t)255;
    return p;
  };
  bf16*  xb   = (bf16*)alloc((size_t)NTOK * DIM * 2);
  bf16*  wqkv = (bf16*)alloc((size_t)3 * DIM * DIM * 2);
  bf16*  wob  = (bf16*)alloc((size_t)DIM * DIM * 2);
  float* bqkv = (float*)alloc((size_t)3 * DIM * 4);
  float* bof  = (float*)alloc((size_t)DIM * 4);
  bf16*  qb   = (bf16*)alloc((size_t)NTOK * DIM * 2);
  bf16*  kb   = (bf16*)alloc((size_t)NTOK * DIM * 2);
  bf16*  vb   = (bf16*)alloc((size_t)NTOK * DIM * 2);
  bf16*  ctx  = (bf16*)alloc((size_t)NTOK * DIM * 2);

  // Fold LoRA: W_eff = W + Bl @ A, to bf16 (QKV stacked)
  prep_weight<<<DIM, 256, 0, stream>>>(wq, laq, lbq, wqkv);
  prep_weight<<<DIM, 256, 0, stream>>>(wk, lak, lbk, wqkv + (size_t)DIM * DIM);
  prep_weight<<<DIM, 256, 0, stream>>>(wv, lav, lbv, wqkv + (size_t)2 * DIM * DIM);
  prep_weight<<<DIM, 256, 0, stream>>>(wo, lao, lbo, wob);
  copyf<<<(DIM + 255) / 256, 256, 0, stream>>>(bq, bqkv, DIM);
  copyf<<<(DIM + 255) / 256, 256, 0, stream>>>(bk, bqkv + DIM, DIM);
  copyf<<<(DIM + 255) / 256, 256, 0, stream>>>(bv, bqkv + 2 * DIM, DIM);
  copyf<<<(DIM + 255) / 256, 256, 0, stream>>>(bo, bof, DIM);
  cvt_x<<<(NTOK * DIM / 8) / 256, 256, 0, stream>>>(x, xb);

  // Fused QKV projection -> per-head bf16 q/k/v
  gemm_bf16<<<(NTOK / 64) * (3 * DIM / 128), 256, 0, stream>>>(
      xb, wqkv, bqkv, 3 * DIM / 128, 0, qb, kb, vb, nullptr);

  // Flash attention
  attn_kernel<<<BATCH * HEADS * (SEQ / 128), 256, 0, stream>>>(qb, kb, vb, ctx);

  // Output projection -> f32
  gemm_bf16<<<(NTOK / 64) * (DIM / 128), 256, 0, stream>>>(
      ctx, wob, bof, DIM / 128, 1, nullptr, nullptr, nullptr, out);
}
